// ASSM_12214886990399
// MI455X (gfx1250) — matrix operand order repaired and checked by reading, not compiled, not run on hardware
//
#include <hip/hip_runtime.h>
#include <hip/hip_bf16.h>
#include <math.h>

// ---------------- problem constants ----------------
#define B_    8
#define C_    192
#define HH    64
#define WW    64
#define NPIX  4096           // H*W
#define D_    384            // HID
#define NST   16
#define NTOK  64
#define IRANK 128
#define RDT   24             // DTR
#define NCH   32             // scan chunks
#define CHL   128            // chunk length (NCH*CHL == NPIX)

typedef __bf16 bf16;
typedef __attribute__((ext_vector_type(16))) __bf16 v16bf;
typedef __attribute__((ext_vector_type(8)))  float  v8f;

// ---------------- CDNA5 async copy to LDS (ASYNCcnt-tracked) ----------------
__device__ __forceinline__ uint32_t lds_addr(const void* p) {
    return (uint32_t)(uintptr_t)p;           // flat addr low 32 bits == LDS offset
}
__device__ __forceinline__ void async_b128(uint32_t lds_off, const void* g) {
    asm volatile("global_load_async_to_lds_b128 %0, %1, off"
                 :: "v"(lds_off), "v"(g) : "memory");
}
__device__ __forceinline__ void async_wait() {
    asm volatile("s_wait_asynccnt 0" ::: "memory");
}

// --------- WMMA fragment loads (contiguous k per lane half, same order for A and B) -----
// A 16x32 bf16: lane holds row, K contiguous: k0 + (hi?16:0) .. +15
__device__ __forceinline__ v16bf frag_a(const bf16* row, int k0, bool hi) {
    v16bf r;
    __builtin_memcpy(&r, row + k0 + (hi ? 16 : 0), 32);
    return r;
}
// B 32x16 bf16: lane holds column, K contiguous: k0 + (hi?16:0) .. +15
__device__ __forceinline__ v16bf frag_b(const bf16* col, int k0, bool hi) {
    v16bf r;
    __builtin_memcpy(&r, col + k0 + (hi ? 16 : 0), 32);
    return r;
}

// ====== 0: full_emb = emb_B@emb_A ; bf16 copies of weight matrices ==========
__global__ void k_prep(const float* __restrict__ embB, const float* __restrict__ embA,
                       const float* __restrict__ ipw, const float* __restrict__ xpw,
                       const float* __restrict__ ow,
                       float* __restrict__ femb, bf16* __restrict__ ipwbf,
                       bf16* __restrict__ xpwbf, bf16* __restrict__ owbf) {
    int gid = blockIdx.x * blockDim.x + threadIdx.x;
    int stride = gridDim.x * blockDim.x;
    for (int e = gid; e < D_ * C_; e += stride) ipwbf[e] = (bf16)ipw[e];
    for (int e = gid; e < 64 * D_; e += stride) {
        int row = e / D_;
        xpwbf[e] = (row < 56) ? (bf16)xpw[e] : (bf16)0.f;    // zero-pad rows 56..63
    }
    for (int e = gid; e < C_ * D_; e += stride) owbf[e] = (bf16)ow[e];
    for (int e = gid; e < NTOK * NST; e += stride) {
        int r = e / NST, nn = e % NST;
        float acc = 0.f;
        #pragma unroll 8
        for (int k = 0; k < IRANK; ++k)
            acc = fmaf(embB[r * IRANK + k], embA[k * NST + nn], acc);
        femb[e] = acc;
    }
}

// ================= 1: routing MLP + gumbel argmax ============================
__global__ void k_route(const float* __restrict__ x, const float* __restrict__ gum,
                        const float* __restrict__ w1, const float* __restrict__ b1,
                        const float* __restrict__ w2, const float* __restrict__ b2,
                        int* __restrict__ idx) {
    __shared__ float h1l[64 * 256];
    int t = threadIdx.x;
    int b = blockIdx.y;
    int l = blockIdx.x * 256 + t;
    const float* xb = x + (size_t)b * C_ * NPIX + l;

    float xc[64];
    for (int kc = 0; kc < 3; ++kc) {
        #pragma unroll
        for (int c = 0; c < 64; ++c) xc[c] = xb[(size_t)(kc * 64 + c) * NPIX];
        for (int o = 0; o < 64; ++o) {
            float acc = (kc == 0) ? b1[o] : h1l[o * 256 + t];
            const float* wr = w1 + o * C_ + kc * 64;
            #pragma unroll
            for (int c = 0; c < 64; ++c) acc = fmaf(xc[c], wr[c], acc);
            h1l[o * 256 + t] = acc;
        }
    }
    float hg[64];
    #pragma unroll
    for (int o = 0; o < 64; ++o) {
        float v = h1l[o * 256 + t];
        hg[o] = 0.5f * v * (1.f + erff(v * 0.70710678118654752f));
    }
    const float* gr = gum + ((size_t)b * NPIX + l) * NTOK;
    float best = -1e30f; int bi = 0;
    for (int o2 = 0; o2 < 64; ++o2) {
        float acc = b2[o2];
        const float* wr = w2 + o2 * 64;
        #pragma unroll
        for (int i = 0; i < 64; ++i) acc = fmaf(hg[i], wr[i], acc);
        acc += gr[o2];
        if (acc > best) { best = acc; bi = o2; }
    }
    idx[(size_t)b * NPIX + l] = bi;
}

// ================= 2: per-batch stable counting sort =========================
__global__ void k_sort(const int* __restrict__ idx, int* __restrict__ srt,
                       int* __restrict__ inv) {
    __shared__ int cnt[NTOK];
    __shared__ int off[NTOK];
    int b = blockIdx.x, t = threadIdx.x;
    if (t < NTOK) cnt[t] = 0;
    __syncthreads();
    for (int l = t; l < NPIX; l += blockDim.x) atomicAdd(&cnt[idx[b * NPIX + l]], 1);
    __syncthreads();
    if (t == 0) {
        int s = 0;
        for (int i = 0; i < NTOK; ++i) { off[i] = s; s += cnt[i]; }
        for (int l = 0; l < NPIX; ++l) {          // stable: ascending l
            int v = idx[b * NPIX + l];
            int p = off[v]++;
            srt[b * NPIX + p] = l;
            inv[b * NPIX + l] = p;
        }
    }
}

// ================= 3: in-proj GEMM via WMMA bf16 =============================
// vraw[b,d,l] = sum_c W[d,c] x[b,c,l] + bias[d]  (32 d-rows x 128 tokens/block)
__global__ void k_inproj(const float* __restrict__ x, const bf16* __restrict__ wbf,
                         const float* __restrict__ pb, float* __restrict__ vraw) {
    __shared__ __align__(32) bf16 wl[32][C_];        // 32 d-rows x 192 k (12KB)
    __shared__ __align__(32) bf16 xsT[128][48];      // [token][k-chunk], padded
    int t = threadIdx.x, wv = t >> 5, L = t & 31;
    int b = blockIdx.z, d0 = blockIdx.y * 32, l0 = blockIdx.x * 128;
    bool hi = L >= 16; int m = L & 15;
    int col = wv * 16 + m;
    int colt = t & 127, rbase = (t >> 7) * 16;       // staging ownership

    // async DMA the 32x192 weight slice (contiguous bf16 rows)
    {
        const char* src = (const char*)(wbf + (size_t)d0 * C_);
        uint32_t dst = lds_addr(&wl[0][0]);
        #pragma unroll
        for (int i = 0; i < 3; ++i) async_b128(dst + t * 16 + i * 4096,
                                               src + t * 16 + i * 4096);
        async_wait();
    }

    v8f acc0, acc1;
    #pragma unroll
    for (int i = 0; i < 8; ++i) { acc0[i] = 0.f; acc1[i] = 0.f; }

    #pragma unroll
    for (int kc = 0; kc < 6; ++kc) {
        int k0 = kc * 32;
        __syncthreads();                              // xsT reuse + wl visibility
        // batched stage: 16 loads in flight -> pack pairs -> 8 ds_store_b32
        {
            const float* src = &x[((size_t)(b * C_ + k0 + rbase)) * NPIX + l0 + colt];
            float v[16];
            #pragma unroll
            for (int i = 0; i < 16; ++i) v[i] = src[(size_t)i * NPIX];
            #pragma unroll
            for (int i = 0; i < 8; ++i) {
                union { uint32_t u; bf16 h[2]; } pk;
                pk.h[0] = (bf16)v[2 * i];
                pk.h[1] = (bf16)v[2 * i + 1];
                *(uint32_t*)&xsT[colt][rbase + 2 * i] = pk.u;
            }
        }
        __syncthreads();
        v16bf bF  = frag_b(&xsT[col][0], 0, hi);
        v16bf aF0 = frag_a(&wl[m][0], k0, hi);
        v16bf aF1 = frag_a(&wl[16 + m][0], k0, hi);
        acc0 = __builtin_amdgcn_wmma_f32_16x16x32_bf16(false, aF0, false, bF,
                                                       (short)0, acc0, false, false);
        acc1 = __builtin_amdgcn_wmma_f32_16x16x32_bf16(false, aF1, false, bF,
                                                       (short)0, acc1, false, false);
    }
    #pragma unroll
    for (int r = 0; r < 8; ++r) {
        int dA = d0 + (hi ? r + 8 : r);
        int dB = dA + 16;
        vraw[((size_t)(b * D_ + dA)) * NPIX + l0 + col] = acc0[r] + pb[dA];
        vraw[((size_t)(b * D_ + dB)) * NPIX + l0 + col] = acc1[r] + pb[dB];
    }
}

// ====== 4: depthwise 3x3 CPE + sigmoid gate + gather into sorted u ===========
__global__ void k_cpe(const float* __restrict__ vraw, const float* __restrict__ cw,
                      const float* __restrict__ cb, const int* __restrict__ inv,
                      float* __restrict__ u, bf16* __restrict__ ubf) {
    int b = blockIdx.z, d = blockIdx.y;
    int l = blockIdx.x * 256 + threadIdx.x;
    const float* vb = vraw + ((size_t)(b * D_ + d)) * NPIX;
    int hh = l >> 6, wp = l & 63;
    float s = 0.f;
    #pragma unroll
    for (int dy = -1; dy <= 1; ++dy) {
        int y2 = hh + dy;
        if ((unsigned)y2 < HH) {
            #pragma unroll
            for (int dx = -1; dx <= 1; ++dx) {
                int x2 = wp + dx;
                if ((unsigned)x2 < WW)
                    s = fmaf(vb[y2 * WW + x2], cw[d * 9 + (dy + 1) * 3 + (dx + 1)], s);
            }
        }
    }
    float cpe = s + cb[d];
    float vg = vb[l] * (1.f / (1.f + expf(-cpe)));
    int j = inv[b * NPIX + l];
    size_t o = ((size_t)(b * NPIX + j)) * D_ + d;
    u[o] = vg;
    ubf[o] = (bf16)vg;
}

// ====== 5: x_dbl via WMMA ; delta = softplus(dt_w@dts + dt_b); Cs=Cm+prompt ==
// 32 tokens per block; each wave: one row-tile x two token tiles (A reuse)
__global__ void k_xdbl(const bf16* __restrict__ ubf, const bf16* __restrict__ xwbf,
                       const float* __restrict__ dtw, const float* __restrict__ dtb,
                       const int* __restrict__ srt, const int* __restrict__ idx,
                       const float* __restrict__ femb,
                       float* __restrict__ dlt, float* __restrict__ Bm,
                       float* __restrict__ Cs) {
    __shared__ __align__(32) bf16 uls[32][D_];       // [token][k] 24KB
    __shared__ __align__(32) bf16 xw[64][D_];        // padded x_proj_w 48KB
    __shared__ float dts[32][RDT];
    int t = threadIdx.x, wv = t >> 5, L = t & 31;
    int b = blockIdx.y, tok0 = blockIdx.x * 32;
    bool hi = L >= 16; int m = L & 15;

    const char* s1 = (const char*)(ubf + ((size_t)(b * NPIX + tok0)) * D_);
    uint32_t d1 = lds_addr(&uls[0][0]);
    #pragma unroll
    for (int i = 0; i < 12; ++i) { int off = t * 16 + i * 2048; async_b128(d1 + off, s1 + off); }
    const char* s2 = (const char*)xwbf;
    uint32_t d2 = lds_addr(&xw[0][0]);
    #pragma unroll
    for (int i = 0; i < 24; ++i) { int off = t * 16 + i * 2048; async_b128(d2 + off, s2 + off); }
    async_wait();
    __syncthreads();

    int r0 = wv * 16;                                // row tile of x_dbl
    v8f acc0, acc1;
    #pragma unroll
    for (int i = 0; i < 8; ++i) { acc0[i] = 0.f; acc1[i] = 0.f; }
    #pragma unroll
    for (int kc = 0; kc < 12; ++kc) {
        v16bf aF = frag_a(&xw[r0 + m][0], kc * 32, hi);
        v16bf bF0 = frag_b(&uls[m][0], kc * 32, hi);
        v16bf bF1 = frag_b(&uls[16 + m][0], kc * 32, hi);
        acc0 = __builtin_amdgcn_wmma_f32_16x16x32_bf16(false, aF, false, bF0,
                                                       (short)0, acc0, false, false);
        acc1 = __builtin_amdgcn_wmma_f32_16x16x32_bf16(false, aF, false, bF1,
                                                       (short)0, acc1, false, false);
    }
    #pragma unroll
    for (int g = 0; g < 2; ++g) {                    // token tile 0 / 1
        int tloc = g * 16 + m;
        int tok = tok0 + tloc;
        int lo = srt[b * NPIX + tok];
        int ti = idx[b * NPIX + lo];
        size_t base = (size_t)(b * NPIX + tok);
        #pragma unroll
        for (int r = 0; r < 8; ++r) {
            int row = r0 + (hi ? r + 8 : r);
            float v = g ? acc1[r] : acc0[r];
            if (row < RDT)            dts[tloc][row] = v;
            else if (row < RDT + NST) Bm[base * NST + (row - RDT)] = v;
            else if (row < 56)        Cs[base * NST + (row - 40)] = v + femb[ti * NST + (row - 40)];
        }
    }
    __syncthreads();
    for (int o = t; o < 32 * D_; o += 128) {
        int tk = o / D_, d = o % D_;
        float acc2 = dtb[d];
        const float* dr = dtw + d * RDT;
        #pragma unroll
        for (int r = 0; r < RDT; ++r) acc2 = fmaf(dts[tk][r], dr[r], acc2);
        float sp = (acc2 > 20.f) ? acc2 : log1pf(expf(acc2));
        dlt[((size_t)(b * NPIX + tok0 + tk)) * D_ + d] = sp;
    }
}

// ====== 6a: scan pass A — per chunk local (P = prod a, Q = local state) ======
__global__ void k_scanA(const float* __restrict__ dlt, const float* __restrict__ u,
                        const float* __restrict__ Bm, const float* __restrict__ Alog,
                        float* __restrict__ Pg, float* __restrict__ Qg) {
    int b = blockIdx.z, ch = blockIdx.y;
    int pair = blockIdx.x * 512 + threadIdx.x;   // pair = d*16 + nn
    int d = pair >> 4, nn = pair & 15;
    float Ad = -expf(Alog[d * NST + nn]);
    float P = 1.f, Q = 0.f;
    int j0 = ch * CHL;
    #pragma unroll 2
    for (int j = j0; j < j0 + CHL; ++j) {
        size_t base = (size_t)(b * NPIX + j);
        float dl = dlt[base * D_ + d];
        float ul = u[base * D_ + d];
        float bl = Bm[base * NST + nn];
        float a = expf(dl * Ad);
        P *= a;
        Q = Q * a + dl * ul * bl;
    }
    size_t o = ((size_t)(b * NCH + ch)) * (D_ * NST) + pair;
    Pg[o] = P; Qg[o] = Q;
}

// ====== 6b: sequential combine across chunks (carry into each chunk) =========
__global__ void k_scanB(const float* __restrict__ Pg, const float* __restrict__ Qg,
                        float* __restrict__ carry) {
    int b = blockIdx.y;
    int pair = blockIdx.x * 512 + threadIdx.x;
    float h = 0.f;
    for (int c = 0; c < NCH; ++c) {
        size_t o = ((size_t)(b * NCH + c)) * (D_ * NST) + pair;
        carry[o] = h;
        h = Pg[o] * h + Qg[o];
    }
}

// ====== 6c: scan pass B — replay chunk from carry, emit y = h.C + Ds*u =======
__global__ void k_scanC(const float* __restrict__ dlt, const float* __restrict__ u,
                        const float* __restrict__ Bm, const float* __restrict__ Cs,
                        const float* __restrict__ Alog, const float* __restrict__ Ds,
                        const float* __restrict__ carry, float* __restrict__ y) {
    int b = blockIdx.z, ch = blockIdx.y;
    int pair = blockIdx.x * 512 + threadIdx.x;
    int d = pair >> 4, nn = pair & 15;
    float Ad = -expf(Alog[d * NST + nn]);
    float Dsd = Ds[d];
    float h = carry[((size_t)(b * NCH + ch)) * (D_ * NST) + pair];
    int j0 = ch * CHL;
    #pragma unroll 2
    for (int j = j0; j < j0 + CHL; ++j) {
        size_t base = (size_t)(b * NPIX + j);
        __builtin_prefetch(&dlt[(base + 1) * D_ + d], 0, 1);
        float dl = dlt[base * D_ + d];
        float ul = u[base * D_ + d];
        float bl = Bm[base * NST + nn];
        float cl = Cs[base * NST + nn];
        float a = expf(dl * Ad);
        h = h * a + dl * ul * bl;
        float contrib = h * cl;
        #pragma unroll
        for (int s = 1; s < 16; s <<= 1) contrib += __shfl_xor(contrib, s, 16);
        if (nn == 0) y[base * D_ + d] = contrib + Dsd * ul;
    }
}

// ====== 7: LayerNorm over D, emit bf16 =======================================
__global__ void k_ln(const float* __restrict__ y, const float* __restrict__ g,
                     const float* __restrict__ be, bf16* __restrict__ yn) {
    int wv = threadIdx.x >> 5, lane = threadIdx.x & 31;
    int b = blockIdx.y, j = blockIdx.x * 8 + wv;
    const float* yr = y + ((size_t)(b * NPIX + j)) * D_;
    float vals[12];
    float s = 0.f;
    #pragma unroll
    for (int i = 0; i < 12; ++i) { vals[i] = yr[lane + 32 * i]; s += vals[i]; }
    #pragma unroll
    for (int o = 16; o >= 1; o >>= 1) s += __shfl_xor(s, o, 32);
    float mu = s * (1.f / D_);
    float v = 0.f;
    #pragma unroll
    for (int i = 0; i < 12; ++i) { float dd = vals[i] - mu; v = fmaf(dd, dd, v); }
    #pragma unroll
    for (int o = 16; o >= 1; o >>= 1) v += __shfl_xor(v, o, 32);
    float rs = rsqrtf(v * (1.f / D_) + 1e-5f);
    bf16* yo = yn + ((size_t)(b * NPIX + j)) * D_;
    #pragma unroll
    for (int i = 0; i < 12; ++i) {
        int k = lane + 32 * i;
        yo[k] = (bf16)((vals[i] - mu) * rs * g[k] + be[k]);
    }
}

// ====== 8: out-proj GEMM via WMMA + bias + inverse-permute scatter ===========
// 32 tokens x 64 c-rows per block; each wave: 2 token tiles sharing B frags
__global__ void k_out(const bf16* __restrict__ yn, const bf16* __restrict__ owbf,
                      const float* __restrict__ ob, const int* __restrict__ srt,
                      float* __restrict__ out) {
    __shared__ __align__(32) bf16 yls[32][D_];       // 24KB (contiguous in global)
    __shared__ __align__(32) bf16 owT[64][D_];       // 48KB c-slice of out_w
    int t = threadIdx.x, wv = t >> 5, L = t & 31;
    int b = blockIdx.z, tok0 = blockIdx.x * 32, cbase = blockIdx.y * 64;
    bool hi = L >= 16; int m = L & 15;

    const char* s1 = (const char*)(yn + ((size_t)(b * NPIX + tok0)) * D_);
    uint32_t d1 = lds_addr(&yls[0][0]);
    #pragma unroll
    for (int i = 0; i < 12; ++i) { int off = t * 16 + i * 2048; async_b128(d1 + off, s1 + off); }
    const char* s2 = (const char*)(owbf + (size_t)cbase * D_);
    uint32_t d2 = lds_addr(&owT[0][0]);
    #pragma unroll
    for (int i = 0; i < 24; ++i) { int off = t * 16 + i * 2048; async_b128(d2 + off, s2 + off); }
    async_wait();
    __syncthreads();

    v8f acc0, acc1;
    #pragma unroll
    for (int i = 0; i < 8; ++i) { acc0[i] = 0.f; acc1[i] = 0.f; }
    int crow = wv * 16 + m;                          // local c row in owT
    #pragma unroll
    for (int kc = 0; kc < 12; ++kc) {
        v16bf bF  = frag_b(&owT[crow][0], kc * 32, hi);
        v16bf aF0 = frag_a(&yls[m][0], kc * 32, hi);
        v16bf aF1 = frag_a(&yls[16 + m][0], kc * 32, hi);
        acc0 = __builtin_amdgcn_wmma_f32_16x16x32_bf16(false, aF0, false, bF,
                                                       (short)0, acc0, false, false);
        acc1 = __builtin_amdgcn_wmma_f32_16x16x32_bf16(false, aF1, false, bF,
                                                       (short)0, acc1, false, false);
    }
    float bias = ob[cbase + crow];
    size_t orow = ((size_t)(b * C_ + (cbase + crow))) * NPIX;
    #pragma unroll
    for (int r = 0; r < 8; ++r) {
        int tokA = tok0 + (hi ? r + 8 : r);
        int tokB = tokA + 16;
        out[orow + srt[b * NPIX + tokA]] = acc0[r] + bias;
        out[orow + srt[b * NPIX + tokB]] = acc1[r] + bias;
    }
}

// ============================ launcher =======================================
extern "C" void kernel_launch(void* const* d_in, const int* in_sizes, int n_in,
                              void* d_out, int out_size, void* d_ws, size_t ws_size,
                              hipStream_t stream) {
    const float* x    = (const float*)d_in[0];
    const float* gum  = (const float*)d_in[1];
    const float* embB = (const float*)d_in[2];
    const float* embA = (const float*)d_in[3];
    const float* rw1  = (const float*)d_in[4];
    const float* rb1  = (const float*)d_in[5];
    const float* rw2  = (const float*)d_in[6];
    const float* rb2  = (const float*)d_in[7];
    const float* ipw  = (const float*)d_in[8];
    const float* ipb  = (const float*)d_in[9];
    const float* cw   = (const float*)d_in[10];
    const float* cb   = (const float*)d_in[11];
    const float* xpw  = (const float*)d_in[12];
    const float* dtw  = (const float*)d_in[13];
    const float* dtb  = (const float*)d_in[14];
    const float* alog = (const float*)d_in[15];
    const float* Ds   = (const float*)d_in[16];
    const float* lng  = (const float*)d_in[17];
    const float* lnb  = (const float*)d_in[18];
    const float* ow   = (const float*)d_in[19];
    const float* ob   = (const float*)d_in[20];
    float* out = (float*)d_out;

    float* ws = (float*)d_ws;
    size_t cur = 0;
    const size_t BN = (size_t)B_ * NPIX;            // 32768 tokens
    const size_t BD = BN * D_;                      // 12.58M

    float* femb  = ws + cur; cur += 1024;
    int*   idx   = (int*)(ws + cur); cur += BN;
    int*   srt   = (int*)(ws + cur); cur += BN;
    int*   inv   = (int*)(ws + cur); cur += BN;
    float* vraw  = ws + cur; cur += BD;             // reused as y after k_cpe
    float* u     = ws + cur; cur += BD;
    float* dlt   = ws + cur; cur += BD;
    float* Bm    = ws + cur; cur += BN * NST;
    float* Cs    = ws + cur; cur += BN * NST;
    float* Pg    = ws + cur; cur += (size_t)B_ * NCH * D_ * NST;
    float* Qg    = ws + cur; cur += (size_t)B_ * NCH * D_ * NST;
    float* carry = ws + cur; cur += (size_t)B_ * NCH * D_ * NST;
    bf16*  yn    = (bf16*)(ws + cur); cur += BD / 2;
    bf16*  ipwbf = (bf16*)(ws + cur); cur += (D_ * C_) / 2;
    bf16*  xpwbf = (bf16*)(ws + cur); cur += (64 * D_) / 2;
    bf16*  owbf  = (bf16*)(ws + cur); cur += (C_ * D_) / 2;
    bf16*  ubf   = (bf16*)(ws + cur); cur += BD / 2;
    float* y     = vraw;                            // alias: vraw dead after k_cpe

    k_prep  <<<256, 256, 0, stream>>>(embB, embA, ipw, xpw, ow, femb, ipwbf, xpwbf, owbf);
    k_route <<<dim3(NPIX / 256, B_), 256, 0, stream>>>(x, gum, rw1, rb1, rw2, rb2, idx);
    k_sort  <<<B_, 256, 0, stream>>>(idx, srt, inv);
    k_inproj<<<dim3(NPIX / 128, D_ / 32, B_), 256, 0, stream>>>(x, ipwbf, ipb, vraw);
    k_cpe   <<<dim3(NPIX / 256, D_, B_), 256, 0, stream>>>(vraw, cw, cb, inv, u, ubf);
    k_xdbl  <<<dim3(NPIX / 32, B_), 128, 0, stream>>>(ubf, xpwbf, dtw, dtb, srt, idx,
                                                      femb, dlt, Bm, Cs);
    k_scanA <<<dim3(D_ * NST / 512, NCH, B_), 512, 0, stream>>>(dlt, u, Bm, alog, Pg, Qg);
    k_scanB <<<dim3(D_ * NST / 512, B_), 512, 0, stream>>>(Pg, Qg, carry);
    k_scanC <<<dim3(D_ * NST / 512, NCH, B_), 512, 0, stream>>>(dlt, u, Bm, Cs, alog,
                                                                Ds, carry, y);
    k_ln    <<<dim3(NPIX / 8, B_), 256, 0, stream>>>(y, lng, lnb, yn);
    k_out   <<<dim3(NPIX / 32, 3, B_), 128, 0, stream>>>(yn, owbf, ob, srt, out);
    (void)in_sizes; (void)n_in; (void)out_size; (void)ws_size;
}
